// CausalSelfAttention_86268713107713
// MI455X (gfx1250) — matrix operand order repaired and checked by reading, not compiled, not run on hardware
//
#include <hip/hip_runtime.h>

typedef __bf16 bf16_t;
typedef __attribute__((ext_vector_type(16))) __bf16 v16bf;
typedef __attribute__((ext_vector_type(8)))  __bf16 v8bf;
typedef __attribute__((ext_vector_type(8)))  float  v8f;
typedef __attribute__((ext_vector_type(4)))  int    v4i;

#define EMBED  1024
#define NHEADS 16
#define HD     64
#define BATCH  4
#define SEQ    2048
#define KQV3   192   // 3*HD

// ---------------------------------------------------------------------------
// WMMA fragment helpers (16x16x32 bf16, wave32; A/B use this code's contiguous-k order)
// A (16x32): lane m=lane&15, half=lane>>4; elems 0..15 -> K = half*16 + i
//            (one contiguous 32B run, loaded as two 16B halves)
// B (32x16): lane col n=lane&15; elems 0..15 -> K = (lane>>4)*16 + i
//            (one contiguous 32B run of a column -> B operands pre-transposed)
// C (16x16 f32): elem r -> row = r + (lane>>4)*8, col = lane&15
// ---------------------------------------------------------------------------
static __device__ inline v16bf load_a_frag(const bf16_t* base, int ld,
                                           int row0, int k0, int lane) {
  const int m    = lane & 15;
  const int half = lane >> 4;
  const bf16_t* p = base + (size_t)(row0 + m) * ld + k0 + half * 16;
  v8bf lo = *(const v8bf*)(p);
  v8bf hi = *(const v8bf*)(p + 8);
  v16bf a;
#pragma unroll
  for (int i = 0; i < 8; ++i) { a[i] = lo[i]; a[i + 8] = hi[i]; }
  return a;
}

static __device__ inline v16bf load_b_frag(const bf16_t* bt, int ld,
                                           int n0, int k0, int lane) {
  const int n  = lane & 15;
  const int kb = (lane >> 4) * 16;
  return *(const v16bf*)(bt + (size_t)(n0 + n) * ld + k0 + kb);
}

static __device__ inline v8f wmma_bf16(v16bf a, v16bf b, v8f c) {
  return __builtin_amdgcn_wmma_f32_16x16x32_bf16(false, a, false, b, (short)0, c,
                                                 false, false);
}

static __device__ inline float red_max16(float v) {
#pragma unroll
  for (int m = 8; m; m >>= 1) v = fmaxf(v, __shfl_xor(v, m, 32));
  return v;
}
static __device__ inline float red_sum16(float v) {
#pragma unroll
  for (int m = 8; m; m >>= 1) v += __shfl_xor(v, m, 32);
  return v;
}

// ---------------------------------------------------------------------------
// gfx1250 async global->LDS copy (ASYNCcnt path) with compile-safe fallback.
// Builtin prototype (from hipcc diagnostic): (AS1 int4*, AS3 int4*, imm, imm)
// ---------------------------------------------------------------------------
static __device__ inline void async_copy16(bf16_t* lds, const bf16_t* g) {
#if __has_builtin(__builtin_amdgcn_global_load_async_to_lds_b128)
  __builtin_amdgcn_global_load_async_to_lds_b128(
      (__attribute__((address_space(1))) v4i*)(g),
      (__attribute__((address_space(3))) v4i*)(lds), 0, 0);
#else
  *(v8bf*)lds = *(const v8bf*)g;   // synchronous fallback
#endif
}
static __device__ inline void wait_async0() {
#if __has_builtin(__builtin_amdgcn_s_wait_asynccnt)
  __builtin_amdgcn_s_wait_asynccnt(0);
#elif __has_builtin(__builtin_amdgcn_global_load_async_to_lds_b128)
  asm volatile("s_wait_asynccnt 0x0" ::: "memory");
#endif
}

// ---------------------------------------------------------------------------
// Precision / layout prep kernels
// ---------------------------------------------------------------------------
__global__ void cvt_f32_bf16(const float* __restrict__ src, bf16_t* __restrict__ dst,
                             int n) {
  int i = blockIdx.x * blockDim.x + threadIdx.x;
  if (i < n) dst[i] = (bf16_t)src[i];
}

// W_kqv [H, D, 192] -> Wt [H, 192, D]  (column-contiguous for B fragments)
__global__ void transpose_wkqv(const float* __restrict__ W, bf16_t* __restrict__ Wt) {
  int d = blockIdx.x * blockDim.x + threadIdx.x;
  int e = blockIdx.y;
  int h = blockIdx.z;
  Wt[((size_t)h * KQV3 + e) * EMBED + d] = (bf16_t)W[((size_t)h * EMBED + d) * KQV3 + e];
}

// W_proj [in, out] -> Wt [out, in]
__global__ void transpose_wproj(const float* __restrict__ W, bf16_t* __restrict__ Wt) {
  int din  = blockIdx.x * blockDim.x + threadIdx.x;
  int dout = blockIdx.y;
  Wt[(size_t)dout * EMBED + din] = (bf16_t)W[(size_t)din * EMBED + dout];
}

// ---------------------------------------------------------------------------
// kqv = x @ W_kqv + b. One wave owns a 64x64 register tile (16 accumulators,
// 16 WMMAs per 32-deep k-step). 192 cols = 3 tiles: tile0->K, tile1->Q,
// tile2->V (written transposed [bh,hd,N] for the P@V B-fragment layout).
// ---------------------------------------------------------------------------
__global__ void __launch_bounds__(256)
kqv_gemm_wmma(const bf16_t* __restrict__ xb, const bf16_t* __restrict__ Wt,
              const float* __restrict__ bias, bf16_t* __restrict__ Kb,
              bf16_t* __restrict__ Qb, bf16_t* __restrict__ Vt) {
  const int lane = threadIdx.x & 31;
  int wave = blockIdx.x * 8 + (threadIdx.x >> 5);
  const int NCOLT = KQV3 / 64;  // 3
  const int NROWT = SEQ / 64;   // 32
  const int colt = wave % NCOLT; wave /= NCOLT;
  const int rowt = wave % NROWT; wave /= NROWT;
  const int bh = wave;
  if (bh >= BATCH * NHEADS) return;
  const int b = bh / NHEADS, h = bh % NHEADS;

  const bf16_t* A  = xb + (size_t)b * SEQ * EMBED;
  const bf16_t* Bt = Wt + (size_t)h * KQV3 * EMBED;
  const int row0 = rowt * 64, n0 = colt * 64;

  v8f acc[4][4] = {};
  for (int k0 = 0; k0 < EMBED; k0 += 32) {
    v16bf a[4], bb[4];
#pragma unroll
    for (int i = 0; i < 4; ++i) a[i]  = load_a_frag(A, EMBED, row0 + 16 * i, k0, lane);
#pragma unroll
    for (int j = 0; j < 4; ++j) bb[j] = load_b_frag(Bt, EMBED, n0 + 16 * j, k0, lane);
#pragma unroll
    for (int i = 0; i < 4; ++i)
#pragma unroll
      for (int j = 0; j < 4; ++j) acc[i][j] = wmma_bf16(a[i], bb[j], acc[i][j]);
  }

  const int n = lane & 15, half = lane >> 4;
#pragma unroll
  for (int j = 0; j < 4; ++j) {
    const int c = n0 + 16 * j + n;           // chunk uniform per wave (colt)
    const float bv = bias[h * KQV3 + c];
#pragma unroll
    for (int i = 0; i < 4; ++i) {
#pragma unroll
      for (int r = 0; r < 8; ++r) {
        const int row = row0 + 16 * i + r + half * 8;
        const float v = acc[i][j][r] + bv;
        if (colt == 0) {
          Kb[((size_t)bh * SEQ + row) * HD + c] = (bf16_t)v;
        } else if (colt == 1) {
          Qb[((size_t)bh * SEQ + row) * HD + (c - HD)] = (bf16_t)v;
        } else {
          Vt[((size_t)bh * HD + (c - 2 * HD)) * SEQ + row] = (bf16_t)v;  // transposed
        }
      }
    }
  }
}

// ---------------------------------------------------------------------------
// Causal flash attention, cooperative: one 256-thread block = 8 waves owning
// 128 consecutive q-rows of one head. 32-key K/V blocks are async-staged into
// double-buffered LDS once per block (ASYNCcnt), shared by all 8 waves.
// Block-uniform kv trip count; fully-masked waves skip math, keep barriers.
// ---------------------------------------------------------------------------
__global__ void __launch_bounds__(256)
flash_attn_wmma(const bf16_t* __restrict__ Qb, const bf16_t* __restrict__ Kb,
                const bf16_t* __restrict__ Vt, bf16_t* __restrict__ sa) {
  __shared__ __align__(64) bf16_t Kl[2][32 * HD];   // 2 x 4KB  [key][hd]
  __shared__ __align__(64) bf16_t Vl[2][HD * 32];   // 2 x 4KB  [hd][key]
  __shared__ __align__(64) bf16_t psh[8][16 * 32];  // per-wave P staging
  const int tid = threadIdx.x;
  const int lane = tid & 31, wslot = tid >> 5;
  const int NQB = SEQ / 128;                        // 16
  const int qb = blockIdx.x % NQB;
  const int bh = blockIdx.x / NQB;
  const int b = bh / NHEADS, h = bh % NHEADS;

  const bf16_t* Qh = Qb + (size_t)bh * SEQ * HD;
  const bf16_t* Kh = Kb + (size_t)bh * SEQ * HD;
  const bf16_t* Vh = Vt + (size_t)bh * HD * SEQ;
  const int Q0 = qb * 128;
  const int q0 = Q0 + wslot * 16;
  const int n = lane & 15, half = lane >> 4;

  const v16bf aq0 = load_a_frag(Qh, HD, q0, 0, lane);
  const v16bf aq1 = load_a_frag(Qh, HD, q0, 32, lane);

  v8f o0 = {}, o1 = {}, o2 = {}, o3 = {};
  float mrow[8], lrow[8];
#pragma unroll
  for (int r = 0; r < 8; ++r) { mrow[r] = -__builtin_inff(); lrow[r] = 0.f; }

  const float scale = 0.125f;  // 1/sqrt(HD)
  const int kvend = Q0 + 128;  // block-uniform causal extent (exclusive)

  // per-thread 16B staging chunks: 256 threads cover each 4KB tile exactly
  const int kkey = tid >> 3, kc8 = (tid & 7) * 8;   // K: [32 keys][64 hd]
  const int vdim = tid >> 2, vk8 = (tid & 3) * 8;   // V: [64 hd][32 keys]

  // prologue: stage buffer 0
  async_copy16(&Kl[0][kkey * HD + kc8], Kh + (size_t)kkey * HD + kc8);
  async_copy16(&Vl[0][vdim * 32 + vk8], Vh + (size_t)vdim * SEQ + vk8);

  int buf = 0;
  for (int kv0 = 0; kv0 < kvend; kv0 += 32, buf ^= 1) {
    wait_async0();
    __syncthreads();                       // buffer `buf` visible to all waves
    if (kv0 + 32 < kvend) {                // stage next block into other buffer
      const int nx = kv0 + 32;
      async_copy16(&Kl[buf ^ 1][kkey * HD + kc8], Kh + (size_t)(nx + kkey) * HD + kc8);
      async_copy16(&Vl[buf ^ 1][vdim * 32 + vk8], Vh + (size_t)vdim * SEQ + nx + vk8);
    }

    if (kv0 <= q0 + 15) {                  // wave not fully masked
      // S = Q @ K^T (16x32): two C tiles, hd reduced over two WMMAs each
      v8f s0 = {}, s1 = {};
      s0 = wmma_bf16(aq0, load_b_frag(&Kl[buf][0], HD, 0, 0, lane), s0);
      s0 = wmma_bf16(aq1, load_b_frag(&Kl[buf][0], HD, 0, 32, lane), s0);
      s1 = wmma_bf16(aq0, load_b_frag(&Kl[buf][0], HD, 16, 0, lane), s1);
      s1 = wmma_bf16(aq1, load_b_frag(&Kl[buf][0], HD, 16, 32, lane), s1);

      // scale + causal mask + online softmax
#pragma unroll
      for (int r = 0; r < 8; ++r) {
        const int rg = q0 + r + half * 8;
        float a = s0[r] * scale, c = s1[r] * scale;
        if (kv0 + n > rg)      a = -__builtin_inff();
        if (kv0 + 16 + n > rg) c = -__builtin_inff();
        const float mx = red_max16(fmaxf(a, c));
        const float mn = fmaxf(mrow[r], mx);
        const float alpha = __expf(mrow[r] - mn);
        const float pa = __expf(a - mn), pc = __expf(c - mn);
        lrow[r] = lrow[r] * alpha + red_sum16(pa + pc);
        mrow[r] = mn;
        o0[r] *= alpha; o1[r] *= alpha; o2[r] *= alpha; o3[r] *= alpha;
        const int prow = r + half * 8;
        psh[wslot][prow * 32 + n]      = (bf16_t)pa;
        psh[wslot][prow * 32 + 16 + n] = (bf16_t)pc;
      }
      __asm__ volatile("" ::: "memory");   // DS in-order per wave; pin order

      // O += P @ V from LDS (V tile already [hd][key] = column-contiguous B)
      const v16bf ap = load_a_frag(&psh[wslot][0], 32, 0, 0, lane);
      o0 = wmma_bf16(ap, load_b_frag(&Vl[buf][0], 32, 0,  0, lane), o0);
      o1 = wmma_bf16(ap, load_b_frag(&Vl[buf][0], 32, 16, 0, lane), o1);
      o2 = wmma_bf16(ap, load_b_frag(&Vl[buf][0], 32, 32, 0, lane), o2);
      o3 = wmma_bf16(ap, load_b_frag(&Vl[buf][0], 32, 48, 0, lane), o3);
      __asm__ volatile("" ::: "memory");
    }
    __syncthreads();                       // all waves done reading `buf`
  }

  // normalize rows, write concat-head activation sa[b, n, h*HD + d] (bf16)
#pragma unroll
  for (int r = 0; r < 8; ++r) {
    const int rg = q0 + r + half * 8;
    const float inv = 1.0f / lrow[r];
    bf16_t* dst = sa + ((size_t)b * SEQ + rg) * EMBED + h * HD;
    dst[n]      = (bf16_t)(o0[r] * inv);
    dst[16 + n] = (bf16_t)(o1[r] * inv);
    dst[32 + n] = (bf16_t)(o2[r] * inv);
    dst[48 + n] = (bf16_t)(o3[r] * inv);
  }
}

// ---------------------------------------------------------------------------
// out = sa @ W_proj + b_proj. 64x64 register tile per wave, fp32 output.
// ---------------------------------------------------------------------------
__global__ void __launch_bounds__(256)
out_proj_wmma(const bf16_t* __restrict__ sa, const bf16_t* __restrict__ WtP,
              const float* __restrict__ bp, float* __restrict__ out) {
  const int lane = threadIdx.x & 31;
  int wave = blockIdx.x * 8 + (threadIdx.x >> 5);
  const int NCOLT = EMBED / 64;            // 16
  const int colt = wave % NCOLT;
  const int rowt = wave / NCOLT;           // 0..127
  if (rowt >= (BATCH * SEQ) / 64) return;
  const int row0 = rowt * 64, n0 = colt * 64;

  v8f acc[4][4] = {};
  for (int k0 = 0; k0 < EMBED; k0 += 32) {
    v16bf a[4], bb[4];
#pragma unroll
    for (int i = 0; i < 4; ++i) a[i]  = load_a_frag(sa, EMBED, row0 + 16 * i, k0, lane);
#pragma unroll
    for (int j = 0; j < 4; ++j) bb[j] = load_b_frag(WtP, EMBED, n0 + 16 * j, k0, lane);
#pragma unroll
    for (int i = 0; i < 4; ++i)
#pragma unroll
      for (int j = 0; j < 4; ++j) acc[i][j] = wmma_bf16(a[i], bb[j], acc[i][j]);
  }

  const int n = lane & 15, half = lane >> 4;
#pragma unroll
  for (int j = 0; j < 4; ++j) {
    const float bv = bp[n0 + 16 * j + n];
#pragma unroll
    for (int i = 0; i < 4; ++i) {
#pragma unroll
      for (int r = 0; r < 8; ++r) {
        out[(size_t)(row0 + 16 * i + r + half * 8) * EMBED + n0 + 16 * j + n] =
            acc[i][j][r] + bv;
      }
    }
  }
}

// ---------------------------------------------------------------------------
extern "C" void kernel_launch(void* const* d_in, const int* in_sizes, int n_in,
                              void* d_out, int out_size, void* d_ws, size_t ws_size,
                              hipStream_t stream) {
  const float* x      = (const float*)d_in[0];
  const float* W_kqv  = (const float*)d_in[1];
  const float* b_kqv  = (const float*)d_in[2];
  const float* W_proj = (const float*)d_in[3];
  const float* b_proj = (const float*)d_in[4];
  float* out = (float*)d_out;

  // workspace carve-out (~92 MB total, bf16)
  bf16_t* xb  = (bf16_t*)d_ws;
  bf16_t* Wtk = xb  + (size_t)BATCH * SEQ * EMBED;
  bf16_t* WtP = Wtk + (size_t)NHEADS * KQV3 * EMBED;
  bf16_t* Kb  = WtP + (size_t)EMBED * EMBED;
  bf16_t* Qb  = Kb  + (size_t)BATCH * NHEADS * SEQ * HD;
  bf16_t* Vt  = Qb  + (size_t)BATCH * NHEADS * SEQ * HD;
  bf16_t* sab = Vt  + (size_t)BATCH * NHEADS * SEQ * HD;

  {
    const int nelem = BATCH * SEQ * EMBED;
    cvt_f32_bf16<<<(nelem + 255) / 256, 256, 0, stream>>>(x, xb, nelem);
  }
  transpose_wkqv<<<dim3(EMBED / 256, KQV3, NHEADS), 256, 0, stream>>>(W_kqv, Wtk);
  transpose_wproj<<<dim3(EMBED / 256, EMBED), 256, 0, stream>>>(W_proj, WtP);

  {
    const int waves = BATCH * NHEADS * (SEQ / 64) * (KQV3 / 64);  // 6144
    kqv_gemm_wmma<<<waves / 8, 256, 0, stream>>>(xb, Wtk, b_kqv, Kb, Qb, Vt);
  }
  {
    const int blocks = BATCH * NHEADS * (SEQ / 128);              // 1024
    flash_attn_wmma<<<blocks, 256, 0, stream>>>(Qb, Kb, Vt, sab);
  }
  {
    const int waves = (BATCH * SEQ / 64) * (EMBED / 64);          // 2048
    out_proj_wmma<<<waves / 8, 256, 0, stream>>>(sab, WtP, b_proj, out);
  }
}
